// GraphSAGE_1047972020370
// MI455X (gfx1250) — compile-verified, matrix operand order checked by reading, not run on hardware
//
#include <hip/hip_runtime.h>

typedef __attribute__((ext_vector_type(16))) _Float16 v16h;
typedef __attribute__((ext_vector_type(8)))  float    v8f;

#define N_NODES     100000
#define N_EDGES     1600000
#define IN_FEATS    128
#define H_FEATS     128
#define NUM_CLASSES 40

// ---------------------------------------------------------------------------
// Degree / inverse-degree
// ---------------------------------------------------------------------------
__global__ void sage_deg_kernel(const int* __restrict__ dst, float* __restrict__ deg) {
    int e = blockIdx.x * blockDim.x + threadIdx.x;
    if (e < N_EDGES) atomicAdd(&deg[dst[e]], 1.0f);
}

__global__ void sage_invdeg_kernel(float* __restrict__ deg) {
    int i = blockIdx.x * blockDim.x + threadIdx.x;
    if (i < N_NODES) {
        float d = deg[i];
        deg[i] = 1.0f / fmaxf(d, 1.0f);
    }
}

// ---------------------------------------------------------------------------
// Edge aggregation: accum[dst] += feat[src]   (32 lanes per edge, float4 each)
// ---------------------------------------------------------------------------
__global__ void sage_agg_kernel(const int* __restrict__ src, const int* __restrict__ dst,
                                const float* __restrict__ feat, float* accum) {
    long long t = (long long)blockIdx.x * blockDim.x + threadIdx.x;
    int e = (int)(t >> 5);
    if (e >= N_EDGES) return;
    int lane = (int)(t & 31);
    const float4 v = *(const float4*)(feat + (long long)src[e] * 128 + lane * 4);
    float* p = accum + (long long)dst[e] * 128 + lane * 4;
    atomicAdd(p + 0, v.x);
    atomicAdd(p + 1, v.y);
    atomicAdd(p + 2, v.z);
    atomicAdd(p + 3, v.w);
}

// ---------------------------------------------------------------------------
// One-shot weight prep: f32 [Wself;Wneigh] (K=256 x NCOLS) -> f16, pre-swizzled
// into WMMA B-fragment order. Linear o = frag*512 + lane*16 + j,
// frag = kc*ntiles + nt.  (j -> k mapping per 16-bit B VGPR layout.)
// ---------------------------------------------------------------------------
__global__ void sage_prep_w_kernel(const float* __restrict__ Wself,
                                   const float* __restrict__ Wneigh,
                                   _Float16* __restrict__ out,
                                   int ntiles, int ncols) {
    int o = blockIdx.x * blockDim.x + threadIdx.x;
    int total = 8 * ntiles * 512;
    if (o >= total) return;
    int frag = o >> 9;
    int lane = (o >> 4) & 31;
    int j    = o & 15;
    int kc   = frag / ntiles;
    int nt   = frag - kc * ntiles;
    int k = kc * 32 + ((j >= 8) ? 16 : 0) + ((lane & 16) ? 8 : 0)
            + ((j >> 1) & 3) * 2 + (j & 1);
    int n = nt * 16 + (lane & 15);
    float w = 0.0f;
    if (n < ncols)
        w = (k < 128) ? Wself[k * ncols + n] : Wneigh[(k - 128) * ncols + n];
    out[o] = (_Float16)w;
}

// ---------------------------------------------------------------------------
// Fused SAGE GEMM:
//   Out[m, n] = act( Aself[m,:]@Wself + (Aneigh[m,:]*inv[m])@Wneigh + bias[n] )
// K = 256 (128 self || 128 neigh), f16 WMMA with f32 accumulate.
// Workgroup: 256 threads (8 waves), 128 rows per workgroup, 16 rows per wave.
// Pre-swizzled f16 weights bulk-copied into LDS with async-to-LDS ops.
// ---------------------------------------------------------------------------
template <int NTILES, bool RELU, int NCOLS>
__global__ __launch_bounds__(256) void sage_gemm_kernel(
    const float* __restrict__ Aself,
    const float* Aneigh,               // may alias Out (row-disjoint per workgroup)
    const float* __restrict__ invdeg,
    const _Float16* __restrict__ Wsw,  // pre-swizzled fragment-order weights
    const float* __restrict__ bias,    // [NCOLS]
    float* Out)                        // [N_NODES, NCOLS]
{
    __shared__ _Float16 sW[8 * NTILES * 512] __attribute__((aligned(32)));

    // ---- async bulk copy: global (pre-swizzled) -> LDS, 16B per lane op ----
    {
        // Low 32 bits of the flat shared address == LDS byte offset
        // (LDS aperture occupies a distinct high-32 range; ISA 10.2).
        const unsigned ldsbase = (unsigned)(unsigned long long)&sW[0];
        const int nchunks = (8 * NTILES * 512 * 2) / 16;
        for (int i = threadIdx.x; i < nchunks; i += 256) {
            unsigned    laddr = ldsbase + i * 16;
            const char* gaddr = (const char*)Wsw + i * 16;
            asm volatile("global_load_async_to_lds_b128 %0, %1, off"
                         :: "v"(laddr), "v"(gaddr)
                         : "memory");
        }
        asm volatile("s_wait_asynccnt 0" ::: "memory");
    }
    __syncthreads();

    const int  tid    = threadIdx.x;
    const int  wave   = tid >> 5;
    const int  lane   = tid & 31;
    const bool hiHalf = (lane & 16) != 0;
    const long long row0  = (long long)blockIdx.x * 128 + wave * 16;
    const long long myrow = row0 + (lane & 15);
    const long long ldrow = (myrow < N_NODES) ? myrow : (long long)(N_NODES - 1);

    const float  inv        = invdeg[ldrow];
    const float* aself_row  = Aself  + ldrow * 128;
    const float* aneigh_row = Aneigh + ldrow * 128;

    v8f acc[NTILES] = {};

    #pragma unroll
    for (int kc = 0; kc < 8; ++kc) {
        // ---- build A fragment (16x32 f16): 8 paired f32 loads per lane ----
        v16h a;
        const int kbase = kc * 32;
        #pragma unroll
        for (int t = 0; t < 8; ++t) {
            int k = kbase + ((t >= 4) ? 16 : 0) + (hiHalf ? 8 : 0) + (t & 3) * 2;
            float2 f2;
            if (k < 128) {
                f2 = *(const float2*)(aself_row + k);
            } else {
                f2 = *(const float2*)(aneigh_row + (k - 128));
                f2.x *= inv;
                f2.y *= inv;
            }
            a[2 * t]     = (_Float16)f2.x;
            a[2 * t + 1] = (_Float16)f2.y;
        }
        // ---- 1 WMMA per column tile; B fragment = contiguous 32B LDS read ----
        #pragma unroll
        for (int nt = 0; nt < NTILES; ++nt) {
            const v16h b = *(const v16h*)(sW + (kc * NTILES + nt) * 512 + lane * 16);
            acc[nt] = __builtin_amdgcn_wmma_f32_16x16x32_f16(
                false, a, false, b, (short)0, acc[nt], false, false);
        }
    }

    // ---- epilogue: bias (+ReLU), predicated store per f32 C/D layout ----
    #pragma unroll
    for (int nt = 0; nt < NTILES; ++nt) {
        int n = nt * 16 + (lane & 15);
        float bn = (n < NCOLS) ? bias[n] : 0.0f;
        #pragma unroll
        for (int r = 0; r < 8; ++r) {
            long long grow = row0 + r + (hiHalf ? 8 : 0);
            if (grow < N_NODES && n < NCOLS) {
                float v = acc[nt][r] + bn;
                if (RELU) v = fmaxf(v, 0.0f);
                Out[grow * NCOLS + n] = v;
            }
        }
    }
}

// ---------------------------------------------------------------------------
extern "C" void kernel_launch(void* const* d_in, const int* in_sizes, int n_in,
                              void* d_out, int out_size, void* d_ws, size_t ws_size,
                              hipStream_t stream) {
    (void)in_sizes; (void)n_in; (void)out_size; (void)ws_size;

    const float* in_feat  = (const float*)d_in[0];
    const int*   src      = (const int*)d_in[1];
    const int*   dst      = (const int*)d_in[2];
    const float* W_self1  = (const float*)d_in[3];
    const float* W_neigh1 = (const float*)d_in[4];
    const float* b1       = (const float*)d_in[5];
    const float* W_self2  = (const float*)d_in[6];
    const float* W_neigh2 = (const float*)d_in[7];
    const float* b2       = (const float*)d_in[8];
    float*       out      = (float*)d_out;

    char*  ws   = (char*)d_ws;
    float* deg  = (float*)ws;
    size_t off  = ((size_t)N_NODES * 4 + 255) & ~(size_t)255;
    float* buf1 = (float*)(ws + off);                    // neigh1 sums, then h (in place)
    float* buf2 = buf1 + (size_t)N_NODES * H_FEATS;      // neigh2 sums
    _Float16* wsw1 = (_Float16*)(buf2 + (size_t)N_NODES * H_FEATS);  // 32768 f16
    _Float16* wsw2 = wsw1 + 8 * 8 * 512;                             // 12288 f16

    // zero accumulators + degree
    hipMemsetAsync(deg,  0, (size_t)N_NODES * sizeof(float), stream);
    hipMemsetAsync(buf1, 0, (size_t)N_NODES * H_FEATS * sizeof(float), stream);
    hipMemsetAsync(buf2, 0, (size_t)N_NODES * H_FEATS * sizeof(float), stream);

    // one-shot: f16 + fragment-swizzle both layers' weights
    sage_prep_w_kernel<<<(8 * 8 * 512 + 255) / 256, 256, 0, stream>>>(
        W_self1, W_neigh1, wsw1, 8, H_FEATS);
    sage_prep_w_kernel<<<(8 * 3 * 512 + 255) / 256, 256, 0, stream>>>(
        W_self2, W_neigh2, wsw2, 3, NUM_CLASSES);

    // degree -> inv_deg (in place)
    sage_deg_kernel<<<(N_EDGES + 255) / 256, 256, 0, stream>>>(dst, deg);
    sage_invdeg_kernel<<<(N_NODES + 255) / 256, 256, 0, stream>>>(deg);

    // layer 1: aggregate, then fused GEMM (self||neigh, K=256) + bias + ReLU
    {
        long long threads = (long long)N_EDGES * 32;
        int blocks = (int)((threads + 255) / 256);
        sage_agg_kernel<<<blocks, 256, 0, stream>>>(src, dst, in_feat, buf1);
    }
    {
        int blocks = (N_NODES + 127) / 128;
        sage_gemm_kernel<8, true, H_FEATS><<<blocks, 256, 0, stream>>>(
            in_feat, buf1, deg, wsw1, b1, buf1 /* h overwrites neigh1 */);
    }

    // layer 2: aggregate h, then fused GEMM (N=40 padded to 48) -> out
    {
        long long threads = (long long)N_EDGES * 32;
        int blocks = (int)((threads + 255) / 256);
        sage_agg_kernel<<<blocks, 256, 0, stream>>>(src, dst, buf1, buf2);
    }
    {
        int blocks = (N_NODES + 127) / 128;
        sage_gemm_kernel<3, false, NUM_CLASSES><<<blocks, 256, 0, stream>>>(
            buf1, buf2, deg, wsw2, b2, out);
    }
}
